// Net_2_78065325572310
// MI455X (gfx1250) — matrix operand order repaired and checked by reading, not compiled, not run on hardware
//
#include <hip/hip_runtime.h>
#include <hip/hip_bf16.h>

typedef __attribute__((ext_vector_type(16))) __bf16 v16bf;
typedef __attribute__((ext_vector_type(8)))  float  v8f;

#define D_DENSE  2048
#define D_SPARSE 32768
#define BATCH    64
#define MROWS    128   // rows 0..63 = x, 64..127 = y

union BF16x16 {
    v16bf  v;
    __bf16 e[16];
    uint4  q[2];
};

// ---------- kernel 1: pack [x;y] fp32 -> bf16 A [128][2048] ----------
__global__ void pack_a_bf16(const float* __restrict__ x, const float* __restrict__ y,
                            __bf16* __restrict__ A) {
    int i = blockIdx.x * blockDim.x + threadIdx.x;
    if (i >= MROWS * D_DENSE) return;
    int row = i / D_DENSE;
    float v = (row < BATCH) ? x[i] : y[i - BATCH * D_DENSE];
    A[i] = (__bf16)v;
}

// ---------- kernel 2: H[128][32768] = A @ W^T  (bf16 WMMA, f32 accum) ----------
// Block: 256 threads = 8 waves. Block owns N-tile of 128 columns, full M=128.
// Wave w: mg = w&1 selects 64-row half, ng = w>>2..  (w>>1) selects 32-col group.
// Each wave: 4 (M) x 2 (N) = 8 accumulator fragments of 16x16.
__global__ __launch_bounds__(256)
void gemm_bf16_wmma(const __bf16* __restrict__ A, const float* __restrict__ W,
                    float* __restrict__ H) {
    const int lane = threadIdx.x & 31;
    const int wave = threadIdx.x >> 5;
    const int mg   = wave & 1;               // 0: rows 0..63, 1: rows 64..127
    const int ng   = wave >> 1;              // 0..3 -> 32-column group
    const int n0   = blockIdx.x * 128 + ng * 32;
    const int lm   = lane & 15;
    const int kh   = lane >> 4;

    v8f zero = {0.f,0.f,0.f,0.f,0.f,0.f,0.f,0.f};
    v8f acc[4][2];
#pragma unroll
    for (int mt = 0; mt < 4; ++mt)
#pragma unroll
        for (int nt = 0; nt < 2; ++nt)
            acc[mt][nt] = zero;

    for (int k0 = 0; k0 < D_DENSE; k0 += 32) {
        // B fragments: lane holds column n = n0+nt*16+lm, K = k0 + kh*16 + (0..15)
        BF16x16 bfr[2];
#pragma unroll
        for (int nt = 0; nt < 2; ++nt) {
            const float* wp = W + (size_t)(n0 + nt * 16 + lm) * D_DENSE + (k0 + kh * 16);
            float wtmp[16];
            *(float4*)(&wtmp[0])  = *(const float4*)(wp + 0);
            *(float4*)(&wtmp[4])  = *(const float4*)(wp + 4);
            *(float4*)(&wtmp[8])  = *(const float4*)(wp + 8);
            *(float4*)(&wtmp[12]) = *(const float4*)(wp + 12);
            __builtin_prefetch(wp + 64, 0, 0);   // stream ahead (global_prefetch)
#pragma unroll
            for (int i = 0; i < 16; ++i) bfr[nt].e[i] = (__bf16)wtmp[i];
        }
        // A fragments: lane holds row m = base+lm; elems 0..15 at K=k0+kh*16+(0..15)
        // (contiguous K order, matching B)
        BF16x16 afr[4];
#pragma unroll
        for (int mt = 0; mt < 4; ++mt) {
            const __bf16* ap = A + (size_t)(mg * 64 + mt * 16 + lm) * D_DENSE + (k0 + kh * 16);
            afr[mt].q[0] = *(const uint4*)(ap);
            afr[mt].q[1] = *(const uint4*)(ap + 8);
        }
#pragma unroll
        for (int mt = 0; mt < 4; ++mt)
#pragma unroll
            for (int nt = 0; nt < 2; ++nt)
                acc[mt][nt] = __builtin_amdgcn_wmma_f32_16x16x32_bf16(
                    false, afr[mt].v, false, bfr[nt].v,
                    (short)0, acc[mt][nt], false, false);
    }

    // D layout: VGPR r -> M = mbase + r (lanes 0..15) / mbase+8+r handled via kh
#pragma unroll
    for (int mt = 0; mt < 4; ++mt) {
#pragma unroll
        for (int nt = 0; nt < 2; ++nt) {
            int n     = n0 + nt * 16 + lm;
            int mbase = mg * 64 + mt * 16 + kh * 8;
#pragma unroll
            for (int r = 0; r < 8; ++r)
                H[(size_t)(mbase + r) * D_SPARSE + n] = acc[mt][nt][r];
        }
    }
}

// ---------- kernel 3: per-column batch mean + rsqrt(var+eps) ----------
// c in [0, 65536): c>>15 selects x/y half, c&32767 = column. Coalesced over c.
__global__ void bn_stats(const float* __restrict__ H,
                         float* __restrict__ mean, float* __restrict__ rs) {
    int c = blockIdx.x * blockDim.x + threadIdx.x;
    if (c >= 2 * D_SPARSE) return;
    int half = c >> 15, n = c & (D_SPARSE - 1);
    const float* p = H + (size_t)(half * BATCH) * D_SPARSE + n;
    float s = 0.f, ss = 0.f;
#pragma unroll 4
    for (int r = 0; r < BATCH; ++r) {
        float v = p[(size_t)r * D_SPARSE];
        s += v; ss += v * v;
    }
    float m   = s * (1.f / BATCH);
    float var = ss * (1.f / BATCH) - m * m;
    mean[c] = m;
    rs[c]   = rsqrtf(var + 1e-5f);
}

// ---------- kernel 4a: zero the per-row accumulators ----------
__global__ void zero_accum(float* __restrict__ accum) {
    int i = threadIdx.x;
    if (i < BATCH * 3) accum[i] = 0.f;
}

// ---------- kernel 4b: BN -> tanh -> WTA(4) -> partial dot/norms ----------
__global__ __launch_bounds__(256)
void wta_cos_partial(const float* __restrict__ H,
                     const float* __restrict__ mean, const float* __restrict__ rs,
                     const float* __restrict__ gx, const float* __restrict__ bx,
                     const float* __restrict__ gy, const float* __restrict__ by,
                     float* __restrict__ accum) {
    int row = blockIdx.y;                                   // 0..63
    int blk = blockIdx.x * blockDim.x + threadIdx.x;        // 4-wide block id
    float dot = 0.f, nx = 0.f, ny = 0.f;
    if (blk < D_SPARSE / 4) {
        int f = blk * 4;
        float tx[4], ty[4];
        float mxv = -__builtin_inff(), myv = -__builtin_inff();
#pragma unroll
        for (int i = 0; i < 4; ++i) {
            int n = f + i;
            float hx = H[(size_t)row * D_SPARSE + n];
            float hy = H[(size_t)(BATCH + row) * D_SPARSE + n];
            tx[i] = tanhf((hx - mean[n]) * rs[n] * gx[n] + bx[n]);
            ty[i] = tanhf((hy - mean[D_SPARSE + n]) * rs[D_SPARSE + n] * gy[n] + by[n]);
            mxv = fmaxf(mxv, tx[i]);
            myv = fmaxf(myv, ty[i]);
        }
#pragma unroll
        for (int i = 0; i < 4; ++i) {
            float a = (tx[i] == mxv) ? tx[i] : 0.f;   // ties kept, as in reference
            float c = (ty[i] == myv) ? ty[i] : 0.f;
            dot += a * c; nx += a * a; ny += c * c;
        }
    }
    // wave32 reduction
#pragma unroll
    for (int off = 16; off > 0; off >>= 1) {
        dot += __shfl_down(dot, off, 32);
        nx  += __shfl_down(nx,  off, 32);
        ny  += __shfl_down(ny,  off, 32);
    }
    if ((threadIdx.x & 31) == 0) {
        atomicAdd(&accum[row * 3 + 0], dot);
        atomicAdd(&accum[row * 3 + 1], nx);
        atomicAdd(&accum[row * 3 + 2], ny);
    }
}

// ---------- kernel 5: cosine epilogue ----------
__global__ void cosine_final(const float* __restrict__ accum, float* __restrict__ out) {
    int r = threadIdx.x;
    if (r >= BATCH) return;
    float dot = accum[r * 3 + 0];
    float na  = fmaxf(sqrtf(accum[r * 3 + 1]), 1e-8f);
    float nb  = fmaxf(sqrtf(accum[r * 3 + 2]), 1e-8f);
    out[r] = dot / (na * nb);
}

extern "C" void kernel_launch(void* const* d_in, const int* in_sizes, int n_in,
                              void* d_out, int out_size, void* d_ws, size_t ws_size,
                              hipStream_t stream) {
    (void)in_sizes; (void)n_in; (void)out_size; (void)ws_size;
    const float* x  = (const float*)d_in[0];
    const float* y  = (const float*)d_in[1];
    const float* W  = (const float*)d_in[2];
    // d_in[3] = b: cancels exactly under BatchNorm (shift-invariant) -> unused
    const float* gx = (const float*)d_in[4];
    const float* bx = (const float*)d_in[5];
    const float* gy = (const float*)d_in[6];
    const float* by = (const float*)d_in[7];
    float* out = (float*)d_out;

    // workspace layout (~17 MB)
    char*   ws    = (char*)d_ws;
    __bf16* A     = (__bf16*)ws;                                        // 512 KB
    float*  H     = (float*)(ws + (512u << 10));                        // 16 MB
    float*  mean  = (float*)(ws + (512u << 10) + (16u << 20));          // 256 KB
    float*  rs    = (float*)(ws + (768u << 10) + (16u << 20));          // 256 KB
    float*  accum = (float*)(ws + (1024u << 10) + (16u << 20));         // 768 B

    pack_a_bf16<<<(MROWS * D_DENSE + 255) / 256, 256, 0, stream>>>(x, y, A);
    gemm_bf16_wmma<<<D_SPARSE / 128, 256, 0, stream>>>(A, W, H);
    bn_stats<<<(2 * D_SPARSE + 255) / 256, 256, 0, stream>>>(H, mean, rs);
    zero_accum<<<1, 256, 0, stream>>>(accum);
    dim3 grid((D_SPARSE / 4 + 255) / 256, BATCH);
    wta_cos_partial<<<grid, 256, 0, stream>>>(H, mean, rs, gx, bx, gy, by, accum);
    cosine_final<<<1, 64, 0, stream>>>(accum, out);
}
